// LLaMAAttention_1082331758948
// MI455X (gfx1250) — matrix operand order repaired and checked by reading, not compiled, not run on hardware
//
#include <hip/hip_runtime.h>

// ---------------------------------------------------------------------------
// LLaMA attention for MI455X (gfx1250, wave32, WMMA).
// Pipeline: [gemm bf16 Q][gemm bf16 K][gemm bf16 V^T][rope][flash attn][gemm f32 out]
// All matmuls use v_wmma_f32_16x16x32_bf16 with f32 accumulation.
// Flash attention stages K/V tiles into LDS with GLOBAL_LOAD_ASYNC_TO_LDS_B128
// (ASYNCcnt) shared by all 8 waves of the workgroup.
// Workspace (d_ws) layout, bf16 elements (needs >= 64 MB):
//   Qb  : TOK*DM   Kb : TOK*DM   Vt : TOK*DM (per-head transposed)   ctx : TOK*DM
// ---------------------------------------------------------------------------

typedef __bf16 bf16;
typedef __attribute__((ext_vector_type(16))) __bf16 v16bf;
typedef __attribute__((ext_vector_type(8)))  float  v8f;

constexpr int DM  = 2048;      // model dim (= N = K of every GEMM)
constexpr int SEQ = 2048;      // sequence length
constexpr int NB  = 2;         // batch
constexpr int NH  = 16;        // heads
constexpr int DH  = 128;       // head dim
constexpr int TOK = NB * SEQ;  // 4096 rows (GEMM M)

union BF16x16 { uint4 u[2]; v16bf v; };
union BF16x8  { uint4 u;    bf16  hh[8]; };

__device__ inline v8f wmma_bf16(v16bf a, v16bf b, v8f c) {
  return __builtin_amdgcn_wmma_f32_16x16x32_bf16(
      false, a, false, b, (short)0, c, false, false);
}

__device__ inline float hmax16(float v) {
  v = fmaxf(v, __shfl_xor(v, 1));
  v = fmaxf(v, __shfl_xor(v, 2));
  v = fmaxf(v, __shfl_xor(v, 4));
  v = fmaxf(v, __shfl_xor(v, 8));
  return v;
}
__device__ inline float hsum16(float v) {
  v += __shfl_xor(v, 1);
  v += __shfl_xor(v, 2);
  v += __shfl_xor(v, 4);
  v += __shfl_xor(v, 8);
  return v;
}

// Async DMA: copy 16 bytes global -> LDS per lane, tracked by ASYNCcnt.
__device__ inline void async_b128(unsigned lds_off, const void* gptr) {
  asm volatile("global_load_async_to_lds_b128 %0, %1, off"
               :: "v"(lds_off), "v"(gptr) : "memory");
}
__device__ inline void wait_async0() {
  asm volatile("s_wait_asynccnt 0x0" ::: "memory");
}

// Load 16 K-contiguous elements from global (f32 or bf16), emit bf16 into LDS.
template<typename AT>
__device__ inline void ldtile(const AT* __restrict__ src, bf16* __restrict__ dst) {
  if constexpr (sizeof(AT) == 4) {
    union { float4 f4[4]; float f[16]; } in;
    const float4* p = (const float4*)src;
    in.f4[0] = p[0]; in.f4[1] = p[1]; in.f4[2] = p[2]; in.f4[3] = p[3];
    union { uint4 u[2]; bf16 h[16]; } o;
#pragma unroll
    for (int e = 0; e < 16; ++e) o.h[e] = (bf16)in.f[e];
    ((uint4*)dst)[0] = o.u[0];
    ((uint4*)dst)[1] = o.u[1];
  } else {
    const uint4* p = (const uint4*)src;
    ((uint4*)dst)[0] = p[0];
    ((uint4*)dst)[1] = p[1];
  }
}

// ---------------------------------------------------------------------------
// GEMM: out[m,n] = sum_k A[m,k] * W[n,k]   (M=TOK, N=DM, K=DM)
// OUT_MODE 0: bf16 row-major (Q, K)
// OUT_MODE 1: bf16 per-head-transposed V^T: out[((b*NH+h)*DH+d)*SEQ + s]
// OUT_MODE 2: f32 row-major (final projection)
// 256 thr = 8 waves; tile 128x128; waves 4(M) x 2(N); LDS K-step 64.
// ---------------------------------------------------------------------------
template<typename AT, int OUT_MODE>
__global__ __launch_bounds__(256, 2)
void gemm2048(const AT* __restrict__ A, const float* __restrict__ W,
              void* __restrict__ outp)
{
  __shared__ bf16 Asl[128 * 64];   // [row][k]  16 KB
  __shared__ bf16 Bsl[128 * 64];   // [col][k]  16 KB

  const int t    = threadIdx.x;
  const int lane = t & 31;
  const int wv   = t >> 5;
  const int hi   = lane >> 4;
  const int ln   = lane & 15;
  const int wm   = (wv >> 1) * 32;
  const int wn   = (wv & 1) * 64;
  const int m0   = blockIdx.y * 128;
  const int n0   = blockIdx.x * 128;

  v8f acc[2][4];
#pragma unroll
  for (int mt = 0; mt < 2; ++mt)
#pragma unroll
    for (int nt = 0; nt < 4; ++nt)
#pragma unroll
      for (int r = 0; r < 8; ++r) acc[mt][nt][r] = 0.0f;

  for (int kt = 0; kt < DM / 64; ++kt) {
    const int k0 = kt * 64;
#pragma unroll
    for (int i = 0; i < 2; ++i) {
      int idx = t + i * 256;
      int row = idx >> 2;
      int ch  = (idx & 3) << 4;
      ldtile<AT>(A + (size_t)(m0 + row) * DM + k0 + ch, &Asl[row * 64 + ch]);
      ldtile<float>(W + (size_t)(n0 + row) * DM + k0 + ch, &Bsl[row * 64 + ch]);
    }
    if (kt + 1 < DM / 64) {           // gfx1250 global_prefetch_b8 of next tile
      __builtin_prefetch(A + (size_t)(m0 + (t >> 1)) * DM + k0 + 64, 0, 1);
      __builtin_prefetch(W + (size_t)(n0 + (t >> 1)) * DM + k0 + 64, 0, 1);
    }
    __syncthreads();

#pragma unroll
    for (int ks = 0; ks < 2; ++ks) {
      BF16x16 af[2];
#pragma unroll
      for (int mt = 0; mt < 2; ++mt) {
        const bf16* p = &Asl[(wm + mt * 16 + ln) * 64 + ks * 32 + hi * 8];
        af[mt].u[0] = *(const uint4*)p;
        af[mt].u[1] = *(const uint4*)(p + 16);
      }
      BF16x16 bfr[4];
#pragma unroll
      for (int nt = 0; nt < 4; ++nt) {
        const bf16* p = &Bsl[(wn + nt * 16 + ln) * 64 + ks * 32 + hi * 8];
        bfr[nt].u[0] = *(const uint4*)p;
        bfr[nt].u[1] = *(const uint4*)(p + 16);
      }
#pragma unroll
      for (int mt = 0; mt < 2; ++mt)
#pragma unroll
        for (int nt = 0; nt < 4; ++nt)
          acc[mt][nt] = wmma_bf16(af[mt].v, bfr[nt].v, acc[mt][nt]);
    }
    __syncthreads();
  }

  if constexpr (OUT_MODE == 0) {
    bf16* O = (bf16*)outp;
#pragma unroll
    for (int mt = 0; mt < 2; ++mt)
#pragma unroll
      for (int nt = 0; nt < 4; ++nt)
#pragma unroll
        for (int r = 0; r < 8; ++r) {
          int m = m0 + wm + mt * 16 + r + 8 * hi;
          int n = n0 + wn + nt * 16 + ln;
          O[(size_t)m * DM + n] = (bf16)acc[mt][nt][r];
        }
  } else if constexpr (OUT_MODE == 1) {
    bf16* O = (bf16*)outp;
#pragma unroll
    for (int mt = 0; mt < 2; ++mt)
#pragma unroll
      for (int nt = 0; nt < 4; ++nt) {
        int n  = n0 + wn + nt * 16 + ln;
        int hd = n >> 7, dd = n & 127;
        int mb = m0 + wm + mt * 16 + 8 * hi;
        int bb = mb >> 11, s0 = mb & (SEQ - 1);
        BF16x8 pk;
#pragma unroll
        for (int r = 0; r < 8; ++r) pk.hh[r] = (bf16)acc[mt][nt][r];
        *(uint4*)&O[((size_t)((bb * NH + hd) * DH + dd)) * SEQ + s0] = pk.u;
      }
  } else {
    float* O = (float*)outp;
#pragma unroll
    for (int mt = 0; mt < 2; ++mt)
#pragma unroll
      for (int nt = 0; nt < 4; ++nt)
#pragma unroll
        for (int r = 0; r < 8; ++r) {
          int m = m0 + wm + mt * 16 + r + 8 * hi;
          int n = n0 + wn + nt * 16 + ln;
          O[(size_t)m * DM + n] = acc[mt][nt][r];
        }
  }
}

// ---------------------------------------------------------------------------
// RoPE on Q and K (bf16 in-place, f32 math). Folds 1/sqrt(DH) into Q.
// ---------------------------------------------------------------------------
__global__ __launch_bounds__(256)
void rope_kernel(bf16* __restrict__ Qm, bf16* __restrict__ Km,
                 const float* __restrict__ cosb, const float* __restrict__ sinb)
{
  int idx = blockIdx.x * 256 + threadIdx.x;      // TOK*NH*64 total
  int d   = idx & 63;
  int rem = idx >> 6;
  int hd  = rem & (NH - 1);
  int tok = rem >> 4;
  int s   = tok & (SEQ - 1);

  float c1 = cosb[s * DH + d],      s1 = sinb[s * DH + d];
  float c2 = cosb[s * DH + d + 64], s2 = sinb[s * DH + d + 64];
  size_t base = (size_t)tok * DM + hd * DH + d;

  const float scl = 0.08838834764831845f;        // DH^-0.5
  float q1 = (float)Qm[base], q2 = (float)Qm[base + 64];
  Qm[base]      = (bf16)((q1 * c1 - q2 * s1) * scl);
  Qm[base + 64] = (bf16)((q2 * c2 + q1 * s2) * scl);

  float k1 = (float)Km[base], k2 = (float)Km[base + 64];
  Km[base]      = (bf16)(k1 * c1 - k2 * s1);
  Km[base + 64] = (bf16)(k2 * c2 + k1 * s2);
}

// ---------------------------------------------------------------------------
// Causal flash attention. Grid: (SEQ/128, NB*NH). Block: 256 thr = 8 waves.
// Each wave owns 16 query rows (128 per WG); K/V 64-key tiles are staged into
// LDS once per WG via async global->LDS DMA and shared by all 8 waves.
// Q, K: bf16 row-major [token][DM]. Vt: bf16 [(b*NH+h)*DH + d][s].
// ---------------------------------------------------------------------------
__global__ __launch_bounds__(256)
void flash128(const bf16* __restrict__ Qm, const bf16* __restrict__ Km,
              const bf16* __restrict__ Vt, bf16* __restrict__ ctx)
{
  __shared__ bf16 Ksl[64 * 128];      // [key][kd]  16 KB
  __shared__ bf16 Vsl[128 * 64];      // [d][s]     16 KB
  __shared__ bf16 Pl[8][16 * 64];     // per-wave P 16 KB

  const int t    = threadIdx.x;
  const int lane = t & 31;
  const int wv   = t >> 5;
  const int hi   = lane >> 4;
  const int ln   = lane & 15;
  const int qt   = blockIdx.x;                 // 128-row query block
  const int bh   = blockIdx.y;
  const int bb   = bh >> 4;
  const int hd   = bh & (NH - 1);
  const int q0   = qt * 128 + wv * 16;         // wave's first query row

  // Q fragments: A(16x32) x 4 covering DH=128, resident for whole kernel.
  BF16x16 qf[4];
  {
    const bf16* qp = Qm + (size_t)(bb * SEQ + q0 + ln) * DM + hd * DH;
#pragma unroll
    for (int ks = 0; ks < 4; ++ks) {
      int kd = ks * 32 + hi * 8;
      qf[ks].u[0] = *(const uint4*)(qp + kd);
      qf[ks].u[1] = *(const uint4*)(qp + kd + 16);
    }
  }

  v8f o[8];
#pragma unroll
  for (int nt = 0; nt < 8; ++nt)
#pragma unroll
    for (int r = 0; r < 8; ++r) o[nt][r] = 0.0f;
  float mrow[8], lrow[8];
#pragma unroll
  for (int r = 0; r < 8; ++r) { mrow[r] = -3.0e38f; lrow[r] = 0.0f; }

  const float L2E = 1.4426950408889634f;
  const bf16* vbase = Vt + (size_t)bh * DH * SEQ;
  const int jmax = 2 * qt + 1;                 // last 64-key tile (inclusive)

  for (int j = 0; j <= jmax; ++j) {
    // ---- async-DMA stage K tile (64x128) and V tile (128x64) into LDS ----
    {
      const bf16* kgb = Km + (size_t)(bb * SEQ + j * 64) * DM + hd * DH;
#pragma unroll
      for (int i = 0; i < 4; ++i) {            // 1024 16B chunks over 256 thr
        int cc = i * 256 + t;
        int key = cc >> 4, inner = (cc & 15) * 8;
        async_b128((unsigned)(size_t)&Ksl[key * 128 + inner],
                   kgb + (size_t)key * DM + inner);
      }
#pragma unroll
      for (int i = 0; i < 4; ++i) {            // 1024 16B chunks over 256 thr
        int cc = i * 256 + t;
        int d = cc >> 3, inner = (cc & 7) * 8;
        async_b128((unsigned)(size_t)&Vsl[d * 64 + inner],
                   vbase + (size_t)d * SEQ + j * 64 + inner);
      }
    }
    wait_async0();
    __syncthreads();

    // ---- S = Q * K^T from LDS (scaling pre-folded into Q) ----
    v8f sc[4];
#pragma unroll
    for (int nt = 0; nt < 4; ++nt)
#pragma unroll
      for (int r = 0; r < 8; ++r) sc[nt][r] = 0.0f;

#pragma unroll
    for (int nt = 0; nt < 4; ++nt) {
      const bf16* kp = &Ksl[(nt * 16 + ln) * 128 + hi * 8];
#pragma unroll
      for (int ks = 0; ks < 4; ++ks) {
        BF16x16 kf;
        kf.u[0] = *(const uint4*)(kp + ks * 32);
        kf.u[1] = *(const uint4*)(kp + ks * 32 + 16);
        sc[nt] = wmma_bf16(qf[ks].v, kf.v, sc[nt]);
      }
    }

    // ---- causal mask (tiles overlapping the diagonal of this wave) ----
    if ((j + 1) * 64 > q0) {
#pragma unroll
      for (int nt = 0; nt < 4; ++nt)
#pragma unroll
        for (int r = 0; r < 8; ++r) {
          int qi = q0 + r + 8 * hi;
          int ki = j * 64 + nt * 16 + ln;
          if (ki > qi) sc[nt][r] = -1.0e30f;
        }
    }

    // ---- online softmax (row = (r, half-wave); reduce across 16 lanes) ----
    float corr[8];
#pragma unroll
    for (int r = 0; r < 8; ++r) {
      float mx = fmaxf(fmaxf(sc[0][r], sc[1][r]), fmaxf(sc[2][r], sc[3][r]));
      mx = hmax16(mx);
      float mnew = fmaxf(mrow[r], mx);
      corr[r] = exp2f((mrow[r] - mnew) * L2E);
      mrow[r] = mnew;
      float rs = 0.0f;
#pragma unroll
      for (int nt = 0; nt < 4; ++nt) {
        float p = exp2f((sc[nt][r] - mnew) * L2E);
        sc[nt][r] = p;
        rs += p;
      }
      rs = hsum16(rs);
      lrow[r] = lrow[r] * corr[r] + rs;
    }
#pragma unroll
    for (int nt = 0; nt < 8; ++nt)
#pragma unroll
      for (int r = 0; r < 8; ++r) o[nt][r] *= corr[r];

    // ---- P (C layout) -> per-wave LDS -> A-fragment layout ----
#pragma unroll
    for (int nt = 0; nt < 4; ++nt)
#pragma unroll
      for (int r = 0; r < 8; ++r)
        Pl[wv][(r + 8 * hi) * 64 + nt * 16 + ln] = (bf16)sc[nt][r];
    asm volatile("s_wait_dscnt 0" ::: "memory");

    // ---- O += P * V from LDS ----
#pragma unroll
    for (int ks = 0; ks < 2; ++ks) {
      BF16x16 pf;
      const bf16* pp = &Pl[wv][ln * 64 + ks * 32 + hi * 8];
      pf.u[0] = *(const uint4*)pp;
      pf.u[1] = *(const uint4*)(pp + 16);
#pragma unroll
      for (int nt = 0; nt < 8; ++nt) {
        const bf16* vp = &Vsl[(nt * 16 + ln) * 64 + ks * 32 + hi * 8];
        BF16x16 vf;
        vf.u[0] = *(const uint4*)vp;
        vf.u[1] = *(const uint4*)(vp + 16);
        o[nt] = wmma_bf16(pf.v, vf.v, o[nt]);
      }
    }
    __syncthreads();   // all waves done reading Ksl/Vsl before next stage
  }

  // ---- normalize and store ctx (b, s, h*DH+d) ----
#pragma unroll
  for (int r = 0; r < 8; ++r) {
    float inv = 1.0f / lrow[r];
    size_t tok = (size_t)(bb * SEQ + q0 + r + 8 * hi);
#pragma unroll
    for (int nt = 0; nt < 8; ++nt)
      ctx[tok * DM + hd * DH + nt * 16 + ln] = (bf16)(o[nt][r] * inv);
  }
}

// ---------------------------------------------------------------------------
extern "C" void kernel_launch(void* const* d_in, const int* in_sizes, int n_in,
                              void* d_out, int out_size, void* d_ws, size_t ws_size,
                              hipStream_t stream)
{
  (void)in_sizes; (void)n_in; (void)out_size; (void)ws_size;
  const float* x    = (const float*)d_in[0];
  const float* Wq   = (const float*)d_in[1];
  const float* Wk   = (const float*)d_in[2];
  const float* Wv   = (const float*)d_in[3];
  const float* Wo   = (const float*)d_in[4];
  const float* cosb = (const float*)d_in[5];
  const float* sinb = (const float*)d_in[6];
  // d_in[7] = mask: causality computed inline.

  bf16* Qb  = (bf16*)d_ws;
  bf16* Kb  = Qb + (size_t)TOK * DM;
  bf16* Vt  = Kb + (size_t)TOK * DM;
  bf16* ctx = Vt + (size_t)TOK * DM;

  dim3 gGemm(DM / 128, TOK / 128);                 // 16 x 32
  gemm2048<float, 0><<<gGemm, 256, 0, stream>>>(x, Wq, Qb);
  gemm2048<float, 0><<<gGemm, 256, 0, stream>>>(x, Wk, Kb);
  gemm2048<float, 1><<<gGemm, 256, 0, stream>>>(x, Wv, Vt);

  int pairs = TOK * NH * 64;                        // 4,194,304
  rope_kernel<<<pairs / 256, 256, 0, stream>>>(Qb, Kb, cosb, sinb);

  flash128<<<dim3(SEQ / 128, NB * NH), 256, 0, stream>>>(Qb, Kb, Vt, ctx);

  gemm2048<bf16, 2><<<gGemm, 256, 0, stream>>>(ctx, Wo, (float*)d_out);
}
